// Generator_61100204753391
// MI455X (gfx1250) — matrix operand order repaired and checked by reading, not compiled, not run on hardware
//
#include <hip/hip_runtime.h>

// ---------------------------------------------------------------------------
// Types for CDNA5 WMMA (gfx1250, wave32)
// ---------------------------------------------------------------------------
typedef __bf16 bf16_t;
typedef __attribute__((ext_vector_type(16))) __bf16 bf16x16;
typedef __attribute__((ext_vector_type(8)))  float  f32x8;

union FragU {
    uint4   q[2];
    bf16x16 v;
};

// A-fragment (16x32 bf16, MxK): lane L: M=L%16, half=L/16.
// element e -> K = (e/8)*16 + half*8 + (e%8): chunk0 @ +0B (16B), chunk1 @ +32B (16B).
static __device__ inline bf16x16 load_frag_a(const bf16_t* p) {
    FragU u;
    u.q[0] = *(const uint4*)(p);
    u.q[1] = *(const uint4*)(p + 16);   // +32 bytes
    return u.v;
}
// B-fragment (32x16 bf16, KxN) from B^T stored [N x K] row-major:
// lane L: N=L%16, element e -> K = (e/8)*16 + half*8 + (e%8): chunk0 @ +0B, chunk1 @ +32B.
static __device__ inline bf16x16 load_frag_b(const bf16_t* p) {
    FragU u;
    u.q[0] = *(const uint4*)(p);
    u.q[1] = *(const uint4*)(p + 16);   // +32 bytes
    return u.v;
}

static __device__ inline float sigmf(float x) {
    return 1.0f / (1.0f + __expf(-x));
}
static __device__ inline float tanh_fast(float x) {
    float e = __expf(-2.0f * fabsf(x));
    float t = (1.0f - e) / (1.0f + e);
    return (x < 0.0f) ? -t : t;
}

// ---------------------------------------------------------------------------
// Problem constants
// ---------------------------------------------------------------------------
#define BB     256
#define TT     256
#define UU     512
#define G3U    1536          // 3*UU
#define EMBD   128
#define K0PAD  160           // 130 padded to multiple of 32
#define NDPAD  32            // 25 padded
#define H16S   528           // UU + 16 bf16 pad (row stride, 1056B, 16B-aligned)

// ---------------------------------------------------------------------------
// Kernel 1: build X0 = [emb[token] | onehot2(label) | zero-pad] as bf16 [B*T, 160]
// ---------------------------------------------------------------------------
__global__ void build_x0_kernel(const int* __restrict__ inputs,
                                const int* __restrict__ labels,
                                const float* __restrict__ emb,
                                bf16_t* __restrict__ X0) {
    int rbt = blockIdx.x;            // 0 .. B*T-1
    int c   = threadIdx.x;           // 0 .. 159
    int b   = rbt >> 8;              // T = 256
    int tok = inputs[rbt];
    float v;
    if (c < EMBD)            v = emb[tok * EMBD + c];
    else if (c == EMBD)      v = (labels[b] == 0) ? 1.0f : 0.0f;
    else if (c == EMBD + 1)  v = (labels[b] == 1) ? 1.0f : 0.0f;
    else                     v = 0.0f;
    X0[(size_t)rbt * K0PAD + c] = (bf16_t)v;
}

// ---------------------------------------------------------------------------
// Kernel 2: transpose f32 [Kin x Nin] -> bf16 [Nout x Kout] with zero padding
// ---------------------------------------------------------------------------
__global__ void transpose_pad_kernel(const float* __restrict__ in,
                                     bf16_t* __restrict__ out,
                                     int Kin, int Nin, int Kout, int Nout) {
    int i = blockIdx.x * blockDim.x + threadIdx.x;
    int total = Nout * Kout;
    if (i >= total) return;
    int n = i / Kout;
    int k = i - n * Kout;
    float v = (k < Kin && n < Nin) ? in[(size_t)k * Nin + n] : 0.0f;
    out[i] = (bf16_t)v;
}

// ---------------------------------------------------------------------------
// Kernel 3: WMMA GEMM  C[M,N](f32) = A[M,K](bf16) @ Bt[N,K](bf16)^T + bias
//  - block: 256 threads (8 waves); grid.x = M/16; wave tile nt = grid.y*8 + wave
//  - A tile staged in LDS with +16 element row pad
// ---------------------------------------------------------------------------
extern __shared__ char g_smem[];

__global__ void __launch_bounds__(256)
gemm_bf16_wmma(const bf16_t* __restrict__ A,
               const bf16_t* __restrict__ Bt,
               const float* __restrict__ bias,
               float* __restrict__ C,
               int N, int K, int ldc) {
    const int KP  = K + 16;
    bf16_t* As = (bf16_t*)g_smem;                  // [16][KP]
    const int tid  = threadIdx.x;
    const int m0   = blockIdx.x * 16;

    // stage A tile (16 rows x K) -> LDS, uint (2 bf16) granularity
    {
        const unsigned* Au = (const unsigned*)(A + (size_t)m0 * K);
        const int kh = K >> 1;                     // uints per row
        for (int idx = tid; idx < 16 * kh; idx += 256) {
            int r  = idx / kh;
            int c2 = idx - r * kh;
            *((unsigned*)(As + r * KP) + c2) = Au[(size_t)r * kh + c2];
        }
    }
    __syncthreads();

    const int wave = tid >> 5;
    const int lane = tid & 31;
    const int nt   = blockIdx.y * 8 + wave;
    if (nt * 16 >= N) return;                      // wave-uniform

    const int frow = lane & 15;
    const int half = lane >> 4;
    const int K32  = K >> 5;

    f32x8 acc = {};
    const bf16_t* arow = As + frow * KP + half * 8;
    const bf16_t* brow = Bt + ((size_t)(nt * 16 + frow)) * K + half * 8;
    for (int kt = 0; kt < K32; ++kt) {
        bf16x16 af = load_frag_a(arow + kt * 32);
        bf16x16 bf = load_frag_b(brow + kt * 32);
        acc = __builtin_amdgcn_wmma_f32_16x16x32_bf16(false, af, false, bf,
                                                      (short)0, acc, false, false);
    }

    const int n = nt * 16 + frow;
    const float bv = bias ? bias[n] : 0.0f;
#pragma unroll
    for (int r = 0; r < 8; ++r) {
        int m = m0 + r + half * 8;
        C[(size_t)m * ldc + n] = acc[r] + bv;
    }
}

// ---------------------------------------------------------------------------
// Kernel 4: GRU recurrent layer (Keras reset_after), one block = 16 batch rows.
//  Per step: rec = h @ rkT^T (WMMA, 96 N-tiles split over 8 waves), then gates.
//  h kept in LDS as f32 (exact state) + bf16 (WMMA A operand).
// ---------------------------------------------------------------------------
__global__ void __launch_bounds__(256)
gru_rec_kernel(const float* __restrict__ xp,      // [B,T,1536]
               const bf16_t* __restrict__ rkT,    // [1536,512] bf16 (transposed)
               const float* __restrict__ br,      // [1536]
               bf16_t* __restrict__ ys,           // [B,T,512] bf16
               float* __restrict__ s_out) {       // [B,512] f32
    char* smem = g_smem;
    bf16_t* h16 = (bf16_t*)(smem);                         // [16][H16S]   16896 B
    float*  h32 = (float*)(smem + 16896);                  // [16][512]    32768 B
    float*  rec = (float*)(smem + 16896 + 32768);          // [16][1536]   98304 B
    float*  brs = (float*)(smem + 16896 + 32768 + 98304);  // [1536]        6144 B

    const int tid  = threadIdx.x;
    const int wave = tid >> 5;
    const int lane = tid & 31;
    const int frow = lane & 15;
    const int half = lane >> 4;
    const int b0   = blockIdx.x * 16;

    for (int i = tid; i < G3U; i += 256) brs[i] = br[i];
    for (int i = tid; i < 16 * UU; i += 256) h32[i] = 0.0f;
    for (int i = tid; i < 16 * H16S; i += 256) h16[i] = (bf16_t)0.0f;
    __syncthreads();

    for (int t = 0; t < TT; ++t) {
        // ---- Phase A: rec = h @ rk  (h: 16x512 bf16, rk^T: [1536x512]) ----
        bf16x16 afrag[16];
        {
            const bf16_t* ap = h16 + frow * H16S + half * 8;
#pragma unroll
            for (int kt = 0; kt < 16; ++kt)
                afrag[kt] = load_frag_a(ap + kt * 32);
        }
#pragma unroll 1
        for (int j = 0; j < 12; ++j) {
            const int nt = wave + j * 8;                  // 0..95
            f32x8 acc = {};
            const bf16_t* brow = rkT + ((size_t)(nt * 16 + frow)) * UU + half * 8;
#pragma unroll
            for (int kt = 0; kt < 16; ++kt) {
                bf16x16 bf = load_frag_b(brow + kt * 32);
                acc = __builtin_amdgcn_wmma_f32_16x16x32_bf16(false, afrag[kt], false, bf,
                                                              (short)0, acc, false, false);
            }
            const int col = nt * 16 + frow;
            const float brv = brs[col];
#pragma unroll
            for (int r = 0; r < 8; ++r)
                rec[(r + half * 8) * G3U + col] = acc[r] + brv;
        }
        __syncthreads();

        // ---- Phase B: gates + state update ----
        const size_t xrow0 = ((size_t)b0 * TT + t) * G3U;   // + r*TT*G3U
#pragma unroll 1
        for (int it = 0; it < 32; ++it) {
            const int i = tid + (it << 8);                  // 0..8191
            const int r = i >> 9;
            const int c = i & 511;
            const size_t xb = xrow0 + (size_t)r * TT * G3U + c;
            const float xz = xp[xb];
            const float xr = xp[xb + 512];
            const float xh = xp[xb + 1024];
            const float rz = rec[r * G3U + c];
            const float rr = rec[r * G3U + c + 512];
            const float rh = rec[r * G3U + c + 1024];
            const float z  = sigmf(xz + rz);
            const float rg = sigmf(xr + rr);
            const float hh = tanh_fast(xh + rg * rh);
            const float hp = h32[r * UU + c];
            const float hn = z * hp + (1.0f - z) * hh;
            h32[r * UU + c]   = hn;
            h16[r * H16S + c] = (bf16_t)hn;
            ys[((size_t)(b0 + r) * TT + t) * UU + c] = (bf16_t)hn;
            if (t == TT - 1) s_out[(size_t)(b0 + r) * UU + c] = hn;
        }
        __syncthreads();
    }
}

// ---------------------------------------------------------------------------
// Kernel 5: finalize — logits (+bd, +99 pad bonus) and mask
// ---------------------------------------------------------------------------
__global__ void finalize_kernel(const float* __restrict__ logits_pad, // [B*T,32]
                                const float* __restrict__ bd,         // [25]
                                const int* __restrict__ inputs,       // [B*T]
                                float* __restrict__ out_logit,        // [B*T,25]
                                float* __restrict__ out_mask) {       // [B*T]
    int rbt = blockIdx.x;
    int c   = threadIdx.x;
    int tok = inputs[rbt];
    if (c < 25) {
        float v = logits_pad[(size_t)rbt * NDPAD + c] + bd[c];
        if (tok == 0 && c == 0) v += 99.0f;
        out_logit[(size_t)rbt * 25 + c] = v;
    }
    if (c == 0) out_mask[rbt] = (tok != 0) ? 1.0f : 0.0f;
}

// ---------------------------------------------------------------------------
// Host launcher
// ---------------------------------------------------------------------------
extern "C" void kernel_launch(void* const* d_in, const int* in_sizes, int n_in,
                              void* d_out, int out_size, void* d_ws, size_t ws_size,
                              hipStream_t stream) {
    (void)in_sizes; (void)n_in; (void)out_size; (void)ws_size;

    const int*   inputs = (const int*)  d_in[0];
    const int*   labels = (const int*)  d_in[1];
    const float* emb    = (const float*)d_in[2];
    const float* k0     = (const float*)d_in[3];
    const float* rk0    = (const float*)d_in[4];
    const float* bi0    = (const float*)d_in[5];
    const float* br0    = (const float*)d_in[6];
    const float* k1     = (const float*)d_in[7];
    const float* rk1    = (const float*)d_in[8];
    const float* bi1    = (const float*)d_in[9];
    const float* br1    = (const float*)d_in[10];
    const float* Wd     = (const float*)d_in[11];
    const float* bd     = (const float*)d_in[12];

    char* ws = (char*)d_ws;
    const size_t MT = (size_t)BB * TT;                 // 65536 rows
    bf16_t* X0    = (bf16_t*)(ws);                                    // 20,971,520
    bf16_t* k0t   = (bf16_t*)(ws + 20971520ull);                      //    491,520
    bf16_t* rk0t  = (bf16_t*)(ws + 21463040ull);                      //  1,572,864
    bf16_t* k1t   = (bf16_t*)(ws + 23035904ull);                      //  1,572,864
    bf16_t* rk1t  = (bf16_t*)(ws + 24608768ull);                      //  1,572,864
    bf16_t* Wdt   = (bf16_t*)(ws + 26181632ull);                      //     32,768
    float*  xpbuf = (float*) (ws + 26214400ull);                      // 402,653,184 (reused)
    bf16_t* ys0   = (bf16_t*)(ws + 428867584ull);                     //  67,108,864
    bf16_t* ys1   = (bf16_t*)(ws + 495976448ull);                     //  67,108,864
    float*  lpad  = (float*) (ws + 563085312ull);                     //   8,388,608

    float* out       = (float*)d_out;
    float* out_logit = out;                           // 1,638,400
    float* out_s0    = out + 1638400;                 //   131,072
    float* out_s1    = out + 1769472;                 //   131,072
    float* out_mask  = out + 1900544;                 //    65,536

    // 1) embedding + one-hot concat -> bf16
    build_x0_kernel<<<(int)MT, K0PAD, 0, stream>>>(inputs, labels, emb, X0);

    // 2) weight transposes to bf16 [N x K]
    auto tgrid = [](int elems) { return (elems + 255) / 256; };
    transpose_pad_kernel<<<tgrid(G3U * K0PAD), 256, 0, stream>>>(k0,  k0t,  130, G3U, K0PAD, G3U);
    transpose_pad_kernel<<<tgrid(G3U * UU),    256, 0, stream>>>(rk0, rk0t, UU,  G3U, UU,    G3U);
    transpose_pad_kernel<<<tgrid(G3U * UU),    256, 0, stream>>>(k1,  k1t,  UU,  G3U, UU,    G3U);
    transpose_pad_kernel<<<tgrid(G3U * UU),    256, 0, stream>>>(rk1, rk1t, UU,  G3U, UU,    G3U);
    transpose_pad_kernel<<<tgrid(NDPAD * UU),  256, 0, stream>>>(Wd,  Wdt,  UU,  25,  UU,    NDPAD);

    // 3) layer-0 input projection: xp = X0 @ k0 + bi0   [65536 x 1536], K=160
    gemm_bf16_wmma<<<dim3(MT / 16, 12), 256, 16 * (K0PAD + 16) * 2, stream>>>(
        X0, k0t, bi0, xpbuf, G3U, K0PAD, G3U);

    // 4) layer-0 recurrence
    gru_rec_kernel<<<BB / 16, 256, 154112, stream>>>(xpbuf, rk0t, br0, ys0, out_s0);

    // 5) layer-1 input projection: xp = ys0 @ k1 + bi1, K=512
    gemm_bf16_wmma<<<dim3(MT / 16, 12), 256, 16 * (UU + 16) * 2, stream>>>(
        ys0, k1t, bi1, xpbuf, G3U, UU, G3U);

    // 6) layer-1 recurrence
    gru_rec_kernel<<<BB / 16, 256, 154112, stream>>>(xpbuf, rk1t, br1, ys1, out_s1);

    // 7) dense head: lpad = ys1 @ Wd  [65536 x 32], K=512 (bias folded in finalize)
    gemm_bf16_wmma<<<dim3(MT / 16, 1), 256, 16 * (UU + 16) * 2, stream>>>(
        ys1, Wdt, nullptr, lpad, NDPAD, UU, NDPAD);

    // 8) finalize: +bd, pad bonus, mask
    finalize_kernel<<<(int)MT, 32, 0, stream>>>(lpad, bd, inputs, out_logit, out_mask);
}
